// Head_66425964200310
// MI455X (gfx1250) — matrix operand order repaired and checked by reading, not compiled, not run on hardware
//
#include <hip/hip_runtime.h>
#include <hip/hip_bf16.h>

typedef _Float16 half_t;
typedef __attribute__((ext_vector_type(4)))  _Float16 v4h;
typedef __attribute__((ext_vector_type(16))) _Float16 v16h;
typedef __attribute__((ext_vector_type(8)))  float    v8f;

static __device__ inline v8f zero8() {
  v8f z;
#pragma unroll
  for (int i = 0; i < 8; ++i) z[i] = 0.f;
  return z;
}

static __device__ inline v8f wmma_f16(v16h a, v16h b, v8f c) {
  // D = A(16x32 f16) * B(32x16 f16) + C(16x16 f32)
  return __builtin_amdgcn_wmma_f32_16x16x32_f16(
      /*neg_a=*/false, a, /*neg_b=*/false, b,
      /*c_mod=*/(short)0, c, /*reuse_a=*/false, /*reuse_b=*/false);
}

// One 16-byte async copy global->LDS per lane (ASYNCcnt tracked, no VGPR data).
static __device__ inline void async_copy16(void* lds_ptr, const void* gptr) {
  uint32_t l = (uint32_t)(uintptr_t)lds_ptr;   // LDS offset (low 32 bits of generic)
  uint64_t g = (uint64_t)(uintptr_t)gptr;
  asm volatile("global_load_async_to_lds_b128 %0, %1, off"
               :: "v"(l), "v"(g) : "memory");
}
static __device__ inline void async_wait0() {
  asm volatile("s_wait_asynccnt 0x0" ::: "memory");
}
static __device__ inline void async_wait4() {
  asm volatile("s_wait_asynccnt 0x4" ::: "memory");
}
static __device__ inline void async_wait6() {
  asm volatile("s_wait_asynccnt 0x6" ::: "memory");
}

// A-fragment (16x32, f16) from LDS tile with row stride `ld` (halfs).
// lane L: row = L&15; VGPR v: K = ((v>>2)<<4) + (L>>4)*8 + (v&3)*2 (holds K,K+1)
static __device__ inline v16h load_a_frag(const half_t* lds, int ld) {
  const int lane = threadIdx.x & 31;
  const int row  = lane & 15;
  const int kh   = lane >> 4;
  v16h a;
#pragma unroll
  for (int v = 0; v < 8; ++v) {
    int k = ((v >> 2) << 4) + kh * 8 + ((v & 3) << 1);
    a[2 * v]     = lds[row * ld + k];
    a[2 * v + 1] = lds[row * ld + k + 1];
  }
  return a;
}

// B-fragment (32x16, f16) where LDS stores element (k,n) at lds[n*ld + k]
// lane L: n = L&15; K = ((v>>2)<<4) + (L>>4)*8 + (v&3)*2 (holds K,K+1)
static __device__ inline v16h load_b_frag_nk(const half_t* lds, int ld) {
  const int lane = threadIdx.x & 31;
  const int n    = lane & 15;
  const int kh   = lane >> 4;
  v16h b;
#pragma unroll
  for (int v = 0; v < 8; ++v) {
    int k = ((v >> 2) << 4) + kh * 8 + ((v & 3) << 1);
    b[2 * v]     = lds[n * ld + k];
    b[2 * v + 1] = lds[n * ld + k + 1];
  }
  return b;
}

// ---------- Kernel 0: convert W (fp32 [k][n]) -> f16 tile-linear [mat][kt][n][kk] ----------
__global__ __launch_bounds__(256) void wprep_kernel(
    const float* __restrict__ Wq, const float* __restrict__ Wk,
    const float* __restrict__ Wv, half_t* __restrict__ wt) {
  int gid = blockIdx.x * 256 + threadIdx.x;  // 3*32*64*32 = 196608 elements
  int kk  = gid & 31;
  int n   = (gid >> 5) & 63;
  int kt  = (gid >> 11) & 31;
  int mat = gid >> 16;
  const float* W = (mat == 0) ? Wq : ((mat == 1) ? Wk : Wv);
  wt[gid] = (half_t)W[(size_t)(kt * 32 + kk) * 64 + n];
}

// ---------------- Kernel 1: q,k,v = x @ {Wq,Wk,Wv}, f16 outputs ----------------
// 64 rows/block, 128 threads (4 waves); K = 1024 in 32 steps; double-buffered:
// W tiles via async-to-LDS from the prepped layout, x tiles via float4+cvt.
// q pre-scaled by C^-0.5; v stored TRANSPOSED per batch: vh[(b*64 + h)*2048 + t].
__global__ __launch_bounds__(128) void qkv_kernel(
    const float* __restrict__ x, const half_t* __restrict__ wt,
    half_t* __restrict__ qh, half_t* __restrict__ kh, half_t* __restrict__ vh) {
  __shared__ __align__(16) half_t sx[2][64 * 32];     // x tiles  [row][k]   4 KB each
  __shared__ __align__(16) half_t sw[2][3][64 * 32];  // W tiles  [n][k]    12 KB each

  const int t     = threadIdx.x;                      // 0..127
  const int wv    = t >> 5;                           // 0..3
  const int lane  = t & 31;
  const int l15   = lane & 15;
  const int lhalf = lane >> 4;
  const int m0    = blockIdx.x * 64;

  v8f acc[3][4];
#pragma unroll
  for (int m = 0; m < 3; ++m)
#pragma unroll
    for (int n = 0; n < 4; ++n) acc[m][n] = zero8();

  auto stage_x = [&](int k0, int buf) {
#pragma unroll
    for (int j = 0; j < 4; ++j) {            // 512 float4 chunks / 128 threads
      int i4  = t + 128 * j;
      int row = i4 >> 3;
      int kk  = (i4 & 7) * 4;
      float4 f = *(const float4*)&x[(size_t)(m0 + row) * 1024 + k0 + kk];
      v4h h;
      h[0] = (half_t)f.x; h[1] = (half_t)f.y;
      h[2] = (half_t)f.z; h[3] = (half_t)f.w;
      *(v4h*)&sx[buf][i4 * 4] = h;
    }
  };
  auto issue_w = [&](int kt, int buf) {
#pragma unroll
    for (int j = 0; j < 6; ++j) {            // 768 chunks of 16 B / 128 lanes
      int c   = t + 128 * j;
      int mat = c >> 8;
      int r   = c & 255;
      async_copy16((char*)sw[buf][mat] + r * 16,
                   (const char*)(wt + (size_t)(mat * 32 + kt) * 2048) + r * 16);
    }
  };

  stage_x(0, 0);
  issue_w(0, 0);

  for (int kt = 0; kt < 32; ++kt) {
    const int cur = kt & 1;
    __syncthreads();                          // all compute on prev tile done
    const bool has_next = (kt + 1 < 32);
    if (has_next) {
      stage_x((kt + 1) * 32, cur ^ 1);
      issue_w(kt + 1, cur ^ 1);
      async_wait6();                          // tile kt's 6 W-asyncs complete
    } else {
      async_wait0();
    }
    __syncthreads();                          // tile kt visible to all waves

    v16h a = load_a_frag(&sx[cur][wv * 16 * 32], 32);
#pragma unroll
    for (int mat = 0; mat < 3; ++mat) {
#pragma unroll
      for (int nt = 0; nt < 4; ++nt) {
        v16h b = load_b_frag_nk(&sw[cur][mat][nt * 16 * 32], 32);
        acc[mat][nt] = wmma_f16(a, b, acc[mat][nt]);
      }
    }
  }

  const float qscale = 0.03125f;  // C^-0.5 = 1024^-0.5
#pragma unroll
  for (int mat = 0; mat < 3; ++mat) {
    float s = (mat == 0) ? qscale : 1.f;
#pragma unroll
    for (int nt = 0; nt < 4; ++nt)
#pragma unroll
      for (int r = 0; r < 8; ++r) {
        int m   = r + 8 * lhalf;
        int row = m0 + wv * 16 + m;
        int col = nt * 16 + l15;
        half_t val = (half_t)(acc[mat][nt][r] * s);
        if (mat == 0) {
          qh[(size_t)row * 64 + col] = val;
        } else if (mat == 1) {
          kh[(size_t)row * 64 + col] = val;
        } else {
          int bb = row >> 11, tloc = row & 2047;       // v transposed [b][h][t]
          vh[((size_t)bb * 64 + col) * 2048 + tloc] = val;
        }
      }
  }
}

// ---------------- Kernel 2: fused causal flash attention ----------------
// 64 query rows/block (4 waves of 16 rows); 32-key tiles, double-buffered
// async k/v staging pipelined against WMMA + online softmax.
__global__ __launch_bounds__(128) void attn_kernel(
    const half_t* __restrict__ qh, const half_t* __restrict__ kh,
    const half_t* __restrict__ vh, float* __restrict__ out) {
  __shared__ __align__(16) half_t sq[64 * 32 * 2];    // q tile [m][h], 8 KB
  __shared__ __align__(16) half_t sk[2][32 * 64];     // k tiles [s][h]  (B: n=s,k=h)
  __shared__ __align__(16) half_t sv[2][64 * 32];     // v^T tiles [h][s](B: n=h,k=s)
  __shared__ __align__(16) half_t sp[4][16 * 32];     // per-wave P tile [m][s]

  const int t     = threadIdx.x;                      // 0..127
  const int wv    = t >> 5;                           // 0..3
  const int lane  = t & 31;
  const int l15   = lane & 15;
  const int lhalf = lane >> 4;
  const int b     = blockIdx.x >> 5;
  const int qrow0 = (blockIdx.x & 31) * 64;
  const size_t rowbase = (size_t)b * 2048;

  auto issue_kv = [&](int s0, int buf) {
    int c = wv * 32 + lane;                   // 128 of 256 chunks; 2 per lane each
#pragma unroll
    for (int j = 0; j < 2; ++j) {
      int ck = c + j * 128;
      async_copy16((char*)sk[buf] + ck * 16,
                   (const char*)(kh + (rowbase + s0) * 64) + ck * 16);
      int h = ck >> 2, part = ck & 3;
      async_copy16((char*)sv[buf] + ck * 16,
                   (const void*)(vh + ((size_t)b * 64 + h) * 2048 + s0 + part * 8));
    }
  };

  // stage q (8 KB = 512 chunks) then first k/v tile
  {
    const char* src = (const char*)(qh + (rowbase + qrow0) * 64);
#pragma unroll
    for (int j = 0; j < 4; ++j) {
      int c = (wv * 4 + j) * 32 + lane;
      async_copy16((char*)sq + c * 16, src + c * 16);
    }
  }
  issue_kv(0, 0);
  async_wait4();                              // oldest 4 (q chunks) done
  __syncthreads();
  const v16h aq0 = load_a_frag(&sq[wv * 16 * 64], 64);        // h = 0..31
  const v16h aq1 = load_a_frag(&sq[wv * 16 * 64 + 32], 64);   // h = 32..63

  v8f accO[4];
#pragma unroll
  for (int nt = 0; nt < 4; ++nt) accO[nt] = zero8();
  float rmax[8], rsum[8];
#pragma unroll
  for (int r = 0; r < 8; ++r) { rmax[r] = -3e38f; rsum[r] = 0.f; }

  const int qi_base  = qrow0 + wv * 16;
  const int wrow_end = qi_base + 15;
  const int smax     = qrow0 + 64;            // block-uniform loop bound

  for (int s0 = 0; s0 < smax; s0 += 32) {
    const int cur = (s0 >> 5) & 1;
    __syncthreads();                          // all reads of buf cur^1 done
    const bool has_next = (s0 + 32 < smax);
    if (has_next) {
      issue_kv(s0 + 32, cur ^ 1);
      async_wait4();                          // tile s0's 4 asyncs complete
    } else {
      async_wait0();
    }
    __syncthreads();                          // tile s0 visible to all waves

    if (s0 <= wrow_end) {
      // S = q @ k^T  (16x32 tile as two 16x16 accs)
      v8f S0 = zero8(), S1 = zero8();
      S0 = wmma_f16(aq0, load_b_frag_nk(&sk[cur][0 * 64 + 0], 64), S0);
      S0 = wmma_f16(aq1, load_b_frag_nk(&sk[cur][0 * 64 + 32], 64), S0);
      S1 = wmma_f16(aq0, load_b_frag_nk(&sk[cur][16 * 64 + 0], 64), S1);
      S1 = wmma_f16(aq1, load_b_frag_nk(&sk[cur][16 * 64 + 32], 64), S1);

      // causal mask (scale already folded into q)
#pragma unroll
      for (int r = 0; r < 8; ++r) {
        int qi = qi_base + r + 8 * lhalf;
        if (s0 + l15 > qi)      S0[r] = -3e38f;
        if (s0 + 16 + l15 > qi) S1[r] = -3e38f;
      }

      // online softmax: rows live in a 16-lane half-group per VGPR index r
      float alpha[8];
#pragma unroll
      for (int r = 0; r < 8; ++r) {
        float mx = fmaxf(S0[r], S1[r]);
#pragma unroll
        for (int off = 1; off < 16; off <<= 1)
          mx = fmaxf(mx, __shfl_xor(mx, off, 32));
        float mnew = fmaxf(rmax[r], mx);
        alpha[r] = __expf(rmax[r] - mnew);
        rmax[r] = mnew;
        float p0 = __expf(S0[r] - mnew);
        float p1 = __expf(S1[r] - mnew);
        S0[r] = p0; S1[r] = p1;
        float ps = p0 + p1;
#pragma unroll
        for (int off = 1; off < 16; off <<= 1)
          ps += __shfl_xor(ps, off, 32);
        rsum[r] = rsum[r] * alpha[r] + ps;
      }
#pragma unroll
      for (int nt = 0; nt < 4; ++nt)
#pragma unroll
        for (int r = 0; r < 8; ++r) accO[nt][r] *= alpha[r];

      // P (D-layout) -> per-wave LDS -> A-fragment layout
      half_t* spw = sp[wv];
#pragma unroll
      for (int r = 0; r < 8; ++r) {
        int m = r + 8 * lhalf;
        spw[m * 32 + l15]      = (half_t)S0[r];
        spw[m * 32 + 16 + l15] = (half_t)S1[r];
      }
      asm volatile("s_wait_dscnt 0" ::: "memory");  // same-wave LDS RAW guard
      v16h ap = load_a_frag(spw, 32);

      // O += P @ V
#pragma unroll
      for (int nt = 0; nt < 4; ++nt)
        accO[nt] = wmma_f16(ap, load_b_frag_nk(&sv[cur][nt * 16 * 32], 32), accO[nt]);
    }
  }

  // normalize and write fp32 output
#pragma unroll
  for (int r = 0; r < 8; ++r) {
    float inv = 1.f / rsum[r];
    int m = r + 8 * lhalf;
    size_t row = rowbase + qrow0 + wv * 16 + m;
#pragma unroll
    for (int nt = 0; nt < 4; ++nt)
      out[row * 64 + nt * 16 + l15] = accO[nt][r] * inv;
  }
}

extern "C" void kernel_launch(void* const* d_in, const int* in_sizes, int n_in,
                              void* d_out, int out_size, void* d_ws, size_t ws_size,
                              hipStream_t stream) {
  const float* x  = (const float*)d_in[0];
  const float* Wk = (const float*)d_in[1];
  const float* Wq = (const float*)d_in[2];
  const float* Wv = (const float*)d_in[3];
  float* out = (float*)d_out;

  const size_t BT = 8 * 2048;          // 16384 rows
  half_t* qh = (half_t*)d_ws;          // 2 MB, [row][h]
  half_t* kh = qh + BT * 64;           // 2 MB, [row][h]
  half_t* vh = kh + BT * 64;           // 2 MB, transposed [b][h][t]
  half_t* wt = vh + BT * 64;           // 384 KB, [mat][kt][n][kk] f16

  wprep_kernel<<<768, 256, 0, stream>>>(Wq, Wk, Wv, wt);
  qkv_kernel<<<256, 128, 0, stream>>>(x, wt, qh, kh, vh);
  attn_kernel<<<256, 128, 0, stream>>>(qh, kh, vh, out);
}
